// LSTMCell_9131100472067
// MI455X (gfx1250) — compile-verified, matrix operand order checked by reading, not run on hardware
//
#include <hip/hip_runtime.h>
#include <hip/hip_bf16.h>

// ---------------------------------------------------------------------------
// LSTM cell fused GEMM for gfx1250 (MI455X).
//   pre[g] = x @ W_x[g]^T + h @ W_h[g]^T + (b_x[g] + b_h[g])
// One fused GEMM: M=4096 (batch), N=4 gates x 1024, K=2048 (x|h concat).
// Compute-bound (AI ~540 FLOP/B) -> v_wmma_f32_16x16x32_bf16 with bf16x3
// hi/lo split (near-f32 accuracy), f32 accumulation.
//
// Fast path (needs 64MB workspace): pre-pass splits x|h and Wx|Wh into
// bf16 hi/lo planes once; the GEMM then double-buffers LDS tiles with
// GLOBAL_LOAD_ASYNC_TO_LDS_B128 (ASYNCcnt) -- no VGPR round trip, no
// per-block conversion VALU. Fallback path converts in-kernel.
// ---------------------------------------------------------------------------

typedef __attribute__((ext_vector_type(16))) __bf16         v16bf;
typedef __attribute__((ext_vector_type(8)))  float          v8f;
typedef __attribute__((ext_vector_type(8)))  unsigned short v8us;

constexpr int BDIM = 4096;          // batch
constexpr int IDIM = 1024;          // input size
constexpr int HDIM = 1024;          // hidden size
constexpr int KDIM = IDIM + HDIM;   // 2048
constexpr int KC   = 32;            // K chunk (one bf16 WMMA K)
constexpr int NKC  = KDIM / KC;     // 64 chunks
constexpr int TM   = 128;           // block M tile
constexpr int TN   = 64;            // block N tile (per gate)
constexpr int LDA  = 40;            // LDS row stride in halves (32 + 8 pad)
constexpr int ABUF = TM * LDA;          // 5120 halves per A plane
constexpr int BBUF = 4 * TN * LDA;      // 10240 halves per B plane
constexpr int BUFSZ = 2 * ABUF + 2 * BBUF;  // 30720 halves = 60KB per buffer
constexpr size_t PLANE = (size_t)BDIM * KDIM;          // 8M elements
constexpr size_t WS_NEED = 4 * PLANE * sizeof(unsigned short);  // 64MB

__device__ __forceinline__ unsigned short f2bf_rn(float f) {
    unsigned int u = __builtin_bit_cast(unsigned int, f);
    u += 0x7FFFu + ((u >> 16) & 1u);            // round-to-nearest-even
    return (unsigned short)(u >> 16);
}
__device__ __forceinline__ float bf2f(unsigned short s) {
    unsigned int u = ((unsigned int)s) << 16;
    return __builtin_bit_cast(float, u);
}
__device__ __forceinline__ float sigmoidf(float v) {
    return 1.0f / (1.0f + __expf(-v));
}

// A/B bf16 WMMA fragment from LDS. base points at (row, koff) inside the
// 32-wide K chunk. halves[0..7]=K koff+0..7, halves[8..15]=K koff+16..23.
__device__ __forceinline__ v16bf load_frag(const unsigned short* base) {
    union { v16bf v; v8us h[2]; } u;
    u.h[0] = *reinterpret_cast<const v8us*>(base);
    u.h[1] = *reinterpret_cast<const v8us*>(base + 16);
    return u.v;
}

// 16B async copy global->LDS, tracked by ASYNCcnt (gfx1250).
__device__ __forceinline__ void async_cp16(unsigned short* lds,
                                           const unsigned short* g) {
    unsigned lds_off = (unsigned)(uintptr_t)lds;          // addr[31:0] = LDS byte offset
    unsigned long long ga = (unsigned long long)(uintptr_t)g;
    asm volatile("global_load_async_to_lds_b128 %0, %1, off"
                 :: "v"(lds_off), "v"(ga) : "memory");
}

// ---------------------------------------------------------------------------
// Pre-pass: split f32 -> bf16 hi/lo planes. xh: [4096][2048] (x|h concat).
// ---------------------------------------------------------------------------
__global__ __launch_bounds__(256)
void convert_xh(const float* __restrict__ x, const float* __restrict__ h,
                unsigned short* __restrict__ hi, unsigned short* __restrict__ lo) {
    const size_t base = ((size_t)blockIdx.x * 256 + threadIdx.x) * 4;
    const int row = (int)(base >> 11);
    const int col = (int)(base & 2047);
    const float* src = (col < IDIM) ? (x + (size_t)row * IDIM + col)
                                    : (h + (size_t)row * HDIM + (col - IDIM));
    const float4 v = *reinterpret_cast<const float4*>(src);
    const float fe[4] = {v.x, v.y, v.z, v.w};
    #pragma unroll
    for (int j = 0; j < 4; ++j) {
        const unsigned short hb = f2bf_rn(fe[j]);
        hi[base + j] = hb;
        lo[base + j] = f2bf_rn(fe[j] - bf2f(hb));
    }
}

// W planes: row r = g*1024+n, cols = Wx[g][n][:] | Wh[g][n][:]  -> [4096][2048]
__global__ __launch_bounds__(256)
void convert_w(const float* __restrict__ Wx, const float* __restrict__ Wh,
               unsigned short* __restrict__ hi, unsigned short* __restrict__ lo) {
    const size_t base = ((size_t)blockIdx.x * 256 + threadIdx.x) * 4;
    const int row = (int)(base >> 11);          // g*1024 + n
    const int col = (int)(base & 2047);
    const float* src = (col < IDIM) ? (Wx + (size_t)row * IDIM + col)
                                    : (Wh + (size_t)row * HDIM + (col - IDIM));
    const float4 v = *reinterpret_cast<const float4*>(src);
    const float fe[4] = {v.x, v.y, v.z, v.w};
    #pragma unroll
    for (int j = 0; j < 4; ++j) {
        const unsigned short hb = f2bf_rn(fe[j]);
        hi[base + j] = hb;
        lo[base + j] = f2bf_rn(fe[j] - bf2f(hb));
    }
}

// ---------------------------------------------------------------------------
// Fast GEMM: async-LDS double-buffered, pre-split bf16 operands.
// ---------------------------------------------------------------------------
__global__ __launch_bounds__(256)
void lstm_cell_wmma_async(const unsigned short* __restrict__ gA_hi,
                          const unsigned short* __restrict__ gA_lo,
                          const unsigned short* __restrict__ gW_hi,
                          const unsigned short* __restrict__ gW_lo,
                          const float* __restrict__ c,
                          const float* __restrict__ bx,
                          const float* __restrict__ bh,
                          float* __restrict__ out) {
    __shared__ unsigned short smem[2 * BUFSZ];    // 120 KB, double buffered

    const int t     = threadIdx.x;
    const int lane  = t & 31;
    const int wv    = t >> 5;
    const int wm    = wv >> 1;
    const int wn    = wv & 1;
    const int cl    = lane & 15;
    const int koff  = (lane < 16) ? 0 : 8;
    const int nBase = blockIdx.x * TN;
    const int mBase = blockIdx.y * TM;

    v8f acc[4][2][2];
    #pragma unroll
    for (int g = 0; g < 4; ++g)
        #pragma unroll
        for (int mf = 0; mf < 2; ++mf)
            #pragma unroll
            for (int nf = 0; nf < 2; ++nf)
                acc[g][mf][nf] = (v8f)0.0f;

    // Issue one K-chunk's 12 async b128 copies (per thread) into buffer `buf`.
    auto issue_chunk = [&](int k0, unsigned short* buf) {
        unsigned short* sA_hi = buf;
        unsigned short* sA_lo = buf + ABUF;
        unsigned short* sB_hi = buf + 2 * ABUF;
        unsigned short* sB_lo = buf + 2 * ABUF + BBUF;
        #pragma unroll
        for (int i = 0; i < 2; ++i) {             // A: 128 rows x 64B
            const int id  = t + i * 256;
            const int r   = id >> 2;
            const int seg = (id & 3) * 8;         // halves
            const size_t go = (size_t)(mBase + r) * KDIM + k0 + seg;
            async_cp16(sA_hi + r * LDA + seg, gA_hi + go);
            async_cp16(sA_lo + r * LDA + seg, gA_lo + go);
        }
        #pragma unroll
        for (int i = 0; i < 4; ++i) {             // B: 4x64 rows x 64B
            const int id  = t + i * 256;
            const int r   = id >> 2;              // 0..255 = g*64+rn
            const int g   = r >> 6;
            const int rn  = r & 63;
            const int seg = (id & 3) * 8;
            const size_t go = (size_t)(g * HDIM + nBase + rn) * KDIM + k0 + seg;
            async_cp16(sB_hi + r * LDA + seg, gW_hi + go);
            async_cp16(sB_lo + r * LDA + seg, gW_lo + go);
        }
    };

    issue_chunk(0, smem);

    for (int chunk = 0; chunk < NKC; ++chunk) {
        unsigned short* buf = smem + (chunk & 1) * BUFSZ;
        if (chunk + 1 < NKC) {
            // prior barrier guarantees everyone finished reading this buffer
            issue_chunk((chunk + 1) * KC, smem + ((chunk + 1) & 1) * BUFSZ);
            // own chunk's copies are the oldest 12 of <=24 outstanding (in-order)
            asm volatile("s_wait_asynccnt 0xc" ::: "memory");
        } else {
            asm volatile("s_wait_asynccnt 0x0" ::: "memory");
        }
        __syncthreads();    // all waves' copies for this buffer are done

        const unsigned short* sA_hi = buf;
        const unsigned short* sA_lo = buf + ABUF;
        const unsigned short* sB_hi = buf + 2 * ABUF;
        const unsigned short* sB_lo = buf + 2 * ABUF + BBUF;

        v16bf a_hi[2], a_lo[2];
        #pragma unroll
        for (int mf = 0; mf < 2; ++mf) {
            const int offA = (wm * 32 + mf * 16 + cl) * LDA + koff;
            a_hi[mf] = load_frag(sA_hi + offA);
            a_lo[mf] = load_frag(sA_lo + offA);
        }
        #pragma unroll
        for (int g = 0; g < 4; ++g) {
            #pragma unroll
            for (int nf = 0; nf < 2; ++nf) {
                const int offB = (g * TN + wn * 32 + nf * 16 + cl) * LDA + koff;
                const v16bf b_hi = load_frag(sB_hi + offB);
                const v16bf b_lo = load_frag(sB_lo + offB);
                #pragma unroll
                for (int mf = 0; mf < 2; ++mf) {
                    acc[g][mf][nf] = __builtin_amdgcn_wmma_f32_16x16x32_bf16(
                        false, a_hi[mf], false, b_hi, (short)0, acc[g][mf][nf], false, false);
                    acc[g][mf][nf] = __builtin_amdgcn_wmma_f32_16x16x32_bf16(
                        false, a_lo[mf], false, b_hi, (short)0, acc[g][mf][nf], false, false);
                    acc[g][mf][nf] = __builtin_amdgcn_wmma_f32_16x16x32_bf16(
                        false, a_hi[mf], false, b_lo, (short)0, acc[g][mf][nf], false, false);
                }
            }
        }
        __syncthreads();    // safe to overwrite this buffer next round
    }

    // ---- fused LSTM epilogue ----
    const int rsel = (lane < 16) ? 0 : 8;
    float* out_o = out;
    float* out_h = out + (size_t)BDIM * HDIM;
    float* out_c = out + 2 * (size_t)BDIM * HDIM;

    #pragma unroll
    for (int nf = 0; nf < 2; ++nf) {
        const int ncol = nBase + wn * 32 + nf * 16 + cl;
        float bsum[4];
        #pragma unroll
        for (int g = 0; g < 4; ++g)
            bsum[g] = bx[g * HDIM + ncol] + bh[g * HDIM + ncol];
        #pragma unroll
        for (int mf = 0; mf < 2; ++mf) {
            #pragma unroll
            for (int j = 0; j < 8; ++j) {
                const int r = mBase + wm * 32 + mf * 16 + j + rsel;
                const size_t idx = (size_t)r * HDIM + ncol;
                const float iv = sigmoidf(acc[0][mf][nf][j] + bsum[0]);
                const float fv = sigmoidf(acc[1][mf][nf][j] + bsum[1]);
                const float gv = tanhf   (acc[2][mf][nf][j] + bsum[2]);
                const float ov = sigmoidf(acc[3][mf][nf][j] + bsum[3]);
                const float cs = fv * c[idx] + iv * gv;
                out_o[idx] = ov;
                out_h[idx] = ov * tanhf(cs);
                out_c[idx] = cs;
            }
        }
    }
}

// ---------------------------------------------------------------------------
// Fallback (no workspace): converts f32->bf16 hi/lo in-kernel (round-1 path).
// ---------------------------------------------------------------------------
__global__ __launch_bounds__(256)
void lstm_cell_wmma_fallback(const float* __restrict__ x,
                             const float* __restrict__ h,
                             const float* __restrict__ c,
                             const float* __restrict__ Wx,
                             const float* __restrict__ Wh,
                             const float* __restrict__ bx,
                             const float* __restrict__ bh,
                             float* __restrict__ out) {
    __shared__ unsigned short smem[BUFSZ];     // 60 KB
    unsigned short* sA_hi = smem;
    unsigned short* sA_lo = sA_hi + ABUF;
    unsigned short* sB_hi = sA_lo + ABUF;
    unsigned short* sB_lo = sB_hi + BBUF;

    const int t     = threadIdx.x;
    const int lane  = t & 31;
    const int wv    = t >> 5;
    const int wm    = wv >> 1;
    const int wn    = wv & 1;
    const int cl    = lane & 15;
    const int koff  = (lane < 16) ? 0 : 8;
    const int nBase = blockIdx.x * TN;
    const int mBase = blockIdx.y * TM;

    v8f acc[4][2][2];
    #pragma unroll
    for (int g = 0; g < 4; ++g)
        #pragma unroll
        for (int mf = 0; mf < 2; ++mf)
            #pragma unroll
            for (int nf = 0; nf < 2; ++nf)
                acc[g][mf][nf] = (v8f)0.0f;

    for (int k0 = 0; k0 < KDIM; k0 += KC) {
        __syncthreads();
        {
            const float* src = (k0 < IDIM) ? (x + (size_t)mBase * IDIM + k0)
                                           : (h + (size_t)mBase * HDIM + (k0 - IDIM));
            #pragma unroll
            for (int i = 0; i < 4; ++i) {
                const int id = t + i * 256;
                const int r  = id >> 3;
                const int cv = (id & 7) * 4;
                const float4 vv = *reinterpret_cast<const float4*>(src + (size_t)r * 1024 + cv);
                const float fe[4] = {vv.x, vv.y, vv.z, vv.w};
                #pragma unroll
                for (int j = 0; j < 4; ++j) {
                    const unsigned short hi = f2bf_rn(fe[j]);
                    sA_hi[r * LDA + cv + j] = hi;
                    sA_lo[r * LDA + cv + j] = f2bf_rn(fe[j] - bf2f(hi));
                }
            }
        }
        {
            const float* src = (k0 < IDIM) ? (Wx + k0) : (Wh + (k0 - IDIM));
            #pragma unroll
            for (int i = 0; i < 8; ++i) {
                const int id = t + i * 256;
                const int g  = id >> 9;
                const int rn = (id >> 3) & 63;
                const int cv = (id & 7) * 4;
                const float4 vv = *reinterpret_cast<const float4*>(
                    src + ((size_t)g * HDIM + nBase + rn) * 1024 + cv);
                const float fe[4] = {vv.x, vv.y, vv.z, vv.w};
                #pragma unroll
                for (int j = 0; j < 4; ++j) {
                    const unsigned short hi = f2bf_rn(fe[j]);
                    sB_hi[(g * TN + rn) * LDA + cv + j] = hi;
                    sB_lo[(g * TN + rn) * LDA + cv + j] = f2bf_rn(fe[j] - bf2f(hi));
                }
            }
        }
        __syncthreads();

        const int k1 = k0 + KC;
        if (k1 < KDIM) {
            const float* na = (k1 < IDIM) ? (x + (size_t)(mBase + (t & 127)) * IDIM + k1)
                                          : (h + (size_t)(mBase + (t & 127)) * HDIM + (k1 - IDIM));
            __builtin_prefetch(na, 0, 3);
            const int g2 = t >> 6, rn2 = t & 63;
            const float* nb = (k1 < IDIM)
                ? (Wx + ((size_t)g2 * HDIM + nBase + rn2) * IDIM + k1)
                : (Wh + ((size_t)g2 * HDIM + nBase + rn2) * HDIM + (k1 - IDIM));
            __builtin_prefetch(nb, 0, 3);
        }

        v16bf a_hi[2], a_lo[2];
        #pragma unroll
        for (int mf = 0; mf < 2; ++mf) {
            const int offA = (wm * 32 + mf * 16 + cl) * LDA + koff;
            a_hi[mf] = load_frag(sA_hi + offA);
            a_lo[mf] = load_frag(sA_lo + offA);
        }
        #pragma unroll
        for (int g = 0; g < 4; ++g) {
            #pragma unroll
            for (int nf = 0; nf < 2; ++nf) {
                const int offB = (g * TN + wn * 32 + nf * 16 + cl) * LDA + koff;
                const v16bf b_hi = load_frag(sB_hi + offB);
                const v16bf b_lo = load_frag(sB_lo + offB);
                #pragma unroll
                for (int mf = 0; mf < 2; ++mf) {
                    acc[g][mf][nf] = __builtin_amdgcn_wmma_f32_16x16x32_bf16(
                        false, a_hi[mf], false, b_hi, (short)0, acc[g][mf][nf], false, false);
                    acc[g][mf][nf] = __builtin_amdgcn_wmma_f32_16x16x32_bf16(
                        false, a_lo[mf], false, b_hi, (short)0, acc[g][mf][nf], false, false);
                    acc[g][mf][nf] = __builtin_amdgcn_wmma_f32_16x16x32_bf16(
                        false, a_hi[mf], false, b_lo, (short)0, acc[g][mf][nf], false, false);
                }
            }
        }
    }

    const int rsel = (lane < 16) ? 0 : 8;
    float* out_o = out;
    float* out_h = out + (size_t)BDIM * HDIM;
    float* out_c = out + 2 * (size_t)BDIM * HDIM;

    #pragma unroll
    for (int nf = 0; nf < 2; ++nf) {
        const int ncol = nBase + wn * 32 + nf * 16 + cl;
        float bsum[4];
        #pragma unroll
        for (int g = 0; g < 4; ++g)
            bsum[g] = bx[g * HDIM + ncol] + bh[g * HDIM + ncol];
        #pragma unroll
        for (int mf = 0; mf < 2; ++mf) {
            #pragma unroll
            for (int j = 0; j < 8; ++j) {
                const int r = mBase + wm * 32 + mf * 16 + j + rsel;
                const size_t idx = (size_t)r * HDIM + ncol;
                const float iv = sigmoidf(acc[0][mf][nf][j] + bsum[0]);
                const float fv = sigmoidf(acc[1][mf][nf][j] + bsum[1]);
                const float gv = tanhf   (acc[2][mf][nf][j] + bsum[2]);
                const float ov = sigmoidf(acc[3][mf][nf][j] + bsum[3]);
                const float cs = fv * c[idx] + iv * gv;
                out_o[idx] = ov;
                out_h[idx] = ov * tanhf(cs);
                out_c[idx] = cs;
            }
        }
    }
}

extern "C" void kernel_launch(void* const* d_in, const int* in_sizes, int n_in,
                              void* d_out, int out_size, void* d_ws, size_t ws_size,
                              hipStream_t stream) {
    (void)in_sizes; (void)n_in; (void)out_size;
    const float* x  = (const float*)d_in[0];
    const float* h  = (const float*)d_in[1];
    const float* c  = (const float*)d_in[2];
    const float* Wx = (const float*)d_in[3];
    const float* Wh = (const float*)d_in[4];
    const float* bx = (const float*)d_in[5];
    const float* bh = (const float*)d_in[6];
    float* out = (float*)d_out;

    dim3 grid(HDIM / TN, BDIM / TM);   // 16 x 32 = 512 blocks
    dim3 block(256);                   // 8 wave32s

    if (ws_size >= WS_NEED) {
        unsigned short* A_hi = (unsigned short*)d_ws;
        unsigned short* A_lo = A_hi + PLANE;
        unsigned short* W_hi = A_lo + PLANE;
        unsigned short* W_lo = W_hi + PLANE;
        const int convBlocks = (int)(PLANE / (256 * 4));   // 8192
        convert_xh<<<convBlocks, 256, 0, stream>>>(x, h, A_hi, A_lo);
        convert_w <<<convBlocks, 256, 0, stream>>>(Wx, Wh, W_hi, W_lo);
        lstm_cell_wmma_async<<<grid, block, 0, stream>>>(
            A_hi, A_lo, W_hi, W_lo, c, bx, bh, out);
    } else {
        lstm_cell_wmma_fallback<<<grid, block, 0, stream>>>(
            x, h, c, Wx, Wh, bx, bh, out);
    }
}
